// IpaAtom10Denoiser_59940563583344
// MI455X (gfx1250) — matrix operand order repaired and checked by reading, not compiled, not run on hardware
//
#include <hip/hip_runtime.h>

// ---------------------------------------------------------------------------
// CDNA5 / gfx1250 fused MLP+LN embedder (node + edge) using bf16 WMMA.
// D = A(16x32 bf16) x B(32x16 bf16) + C(16x16 f32) via v_wmma_f32_16x16x32_bf16
// Edge kernel: 64 rows/WG, B-fragment loaded once per 4 row-tiles (4x less L2).
// ---------------------------------------------------------------------------

typedef __attribute__((ext_vector_type(16))) __bf16 v16bf;
typedef __attribute__((ext_vector_type(8)))  float  v8f;

union BF16Frag { unsigned u[8]; v16bf v; };

__device__ __forceinline__ unsigned short f32_bf16(float f) {
  unsigned u = __float_as_uint(f);
  u += 0x7fffu + ((u >> 16) & 1u);   // round to nearest even
  return (unsigned short)(u >> 16);
}

// A-fragment (16x32 bf16, M x K) from LDS row-major X (u32 view, sw = row
// stride in dwords).  ISA layout: lane L: m = L&15, kh = L>>4;
// VGPR v holds K pair k0 = c*32 + kh*8 + (v&3)*2 + (v>>2)*16.
__device__ __forceinline__ v16bf load_a(const unsigned* Xw, int m, int kh, int c, int sw) {
  BF16Frag r;
  int base = m * sw + c * 16 + kh * 4;
#pragma unroll
  for (int v = 0; v < 4; ++v) { r.u[v] = Xw[base + v]; r.u[v + 4] = Xw[base + 8 + v]; }
  return r.v;
}

// B-fragment pre-swizzled in ws: index ((t*KC+c)*32 + lane)*8 + v
__device__ __forceinline__ v16bf load_b(const unsigned* __restrict__ Wf, int tc, int lane) {
  BF16Frag r;
  const unsigned* p = Wf + ((size_t)tc * 32 + lane) * 8;
#pragma unroll
  for (int v = 0; v < 8; ++v) r.u[v] = p[v];
  return r.v;
}

// One 16-row tile through a layer (used by node kernel).
template<int KC>
__device__ __forceinline__ v8f mm16(const unsigned* Xw, int sw, const unsigned* __restrict__ Wf,
                                    int t, int m, int kh, int lane) {
  v8f acc = {0.f, 0.f, 0.f, 0.f, 0.f, 0.f, 0.f, 0.f};
#pragma unroll
  for (int c = 0; c < KC; ++c) {
    v16bf a = load_a(Xw, m, kh, c, sw);
    v16bf b = load_b(Wf, t * KC + c, lane);
    acc = __builtin_amdgcn_wmma_f32_16x16x32_bf16(false, a, false, b, (short)0, acc,
                                                  false, false);
  }
  return acc;
}

// Four 16-row tiles through a layer sharing each B fragment (edge kernel).
template<int KC>
__device__ __forceinline__ void mm64(const unsigned* Xw, int sw, const unsigned* __restrict__ Wf,
                                     int t, int m, int kh, int lane, v8f acc[4]) {
#pragma unroll
  for (int c = 0; c < KC; ++c) {
    v16bf b = load_b(Wf, t * KC + c, lane);
#pragma unroll
    for (int mt = 0; mt < 4; ++mt) {
      v16bf a = load_a(Xw + mt * 16 * sw, m, kh, c, sw);
      acc[mt] = __builtin_amdgcn_wmma_f32_16x16x32_bf16(false, a, false, b, (short)0,
                                                        acc[mt], false, false);
    }
  }
}

// Store C/D tile (lane: n = t*16 + (lane&15); VGPR v: m = rowbase + kh*8 + v)
// as bf16 into LDS row-major H (se = row stride in bf16 elements).
__device__ __forceinline__ void store_h(unsigned short* H, int se, int rowbase, int t,
                                        int kh, int lane, const v8f& h) {
  int n = t * 16 + (lane & 15);
#pragma unroll
  for (int v = 0; v < 8; ++v) H[(rowbase + kh * 8 + v) * se + n] = f32_bf16(h[v]);
}

// ---------------------------------------------------------------------------
// Prep kernels
// ---------------------------------------------------------------------------

// prot_t_embed[b,n,73] -> bf16, row stride 80
__global__ __launch_bounds__(256) void prep_prot73_k(
    const float* __restrict__ noised, const float* __restrict__ tval,
    const float* __restrict__ fm, unsigned short* __restrict__ o73) {
  int r = blockIdx.x * 256 + threadIdx.x;
  if (r >= 768) return;
  int b = r / 384;
  unsigned short* o = o73 + r * 80;
  float tv = tval[b] * 10000.f;
#pragma unroll
  for (int k = 0; k < 16; ++k) {
    float fr = expf(-(9.210340371976184f / 15.f) * (float)k);  // ln(10000)
    float ang = tv * fr;
    o[k]      = f32_bf16(sinf(ang));
    o[16 + k] = f32_bf16(cosf(ang));
  }
  o[32] = f32_bf16(fm[r]);
  const float* na = noised + r * 30;
#pragma unroll
  for (int q = 0; q < 30; ++q) o[33 + q] = f32_bf16(na[q]);
#pragma unroll
  for (int a = 0; a < 10; ++a) {
    float x = na[3 * a], y = na[3 * a + 1], z = na[3 * a + 2];
    o[63 + a] = f32_bf16(1.f / (1.f + x * x + y * y + z * z));
  }
#pragma unroll
  for (int q = 73; q < 80; ++q) o[q] = 0;
}

// Convert W(K x C, f32) -> bf16 B-fragments, K padded to KC*32 with zeros.
__global__ __launch_bounds__(256) void prep_frags_k(
    const float* __restrict__ W, int K, int C, int KC,
    unsigned* __restrict__ out, int total) {
  int idx = blockIdx.x * 256 + threadIdx.x;
  if (idx >= total) return;
  int v = idx & 7;
  int lane = (idx >> 3) & 31;
  int tc = idx >> 8;                // t*KC + c
  int c = tc % KC;
  int t = tc / KC;
  int n = t * 16 + (lane & 15);
  int k = c * 32 + (lane >> 4) * 8 + (v & 3) * 2 + (v >> 2) * 16;
  float lo = (k < K)     ? W[(size_t)k * C + n]       : 0.f;
  float hi = (k + 1 < K) ? W[(size_t)(k + 1) * C + n] : 0.f;
  out[idx] = (unsigned)f32_bf16(lo) | ((unsigned)f32_bf16(hi) << 16);
}

// ---------------------------------------------------------------------------
// Edge MLP: rows = B*N*N, EDGE_IN=200 (pad 224) -> 128 -> 128 -> 128 + LN
// 64 rows per workgroup; 8 waves, wave = 16-col n-tile.
// ---------------------------------------------------------------------------
__global__ __launch_bounds__(256) void edge_mlp(
    const unsigned short* __restrict__ prot73, const int* __restrict__ seq,
    const float* __restrict__ trans,
    const unsigned* __restrict__ w1f, const unsigned* __restrict__ w2f,
    const unsigned* __restrict__ w3f,
    const float* __restrict__ b1, const float* __restrict__ b2,
    const float* __restrict__ b3,
    const float* __restrict__ g, const float* __restrict__ bt,
    float* __restrict__ out) {
  constexpr int SXW = 114;   // X row stride in dwords (228 bf16)
  constexpr int SHW = 66;    // H row stride in dwords (132 bf16)
  constexpr int SH3 = 132;   // H3 row stride in f32
  // SMEM regions: [0, 33792) X (64x228 bf16 = 29184 B, padded) aliased with
  // H3 (64x132 f32 = 33792 B, X dead after layer1); then two H buffers.
  __shared__ __align__(16) char SMEM[33792 + 2 * 16896];
  unsigned* Xw  = (unsigned*)SMEM;
  float*    H3  = (float*)SMEM;
  unsigned* Hw0 = (unsigned*)(SMEM + 33792);
  unsigned* Hw1 = (unsigned*)(SMEM + 33792 + 16896);
  unsigned short* Xs = (unsigned short*)SMEM;

  const int tid = threadIdx.x;
  const int r0  = blockIdx.x * 64;
  const int b   = r0 / (384 * 384);
  const int rem = r0 - b * 384 * 384;
  const int i   = rem / 384;
  const int j0  = rem - i * 384;     // rows share b,i; j = j0..j0+63

  const int   si  = seq[b * 384 + i];
  const float tix = trans[(b * 384 + i) * 3 + 0];
  const float tiy = trans[(b * 384 + i) * 3 + 1];
  const float tiz = trans[(b * 384 + i) * 3 + 2];

  // cols 0..72 = prot73[i] (broadcast), 73..145 = prot73[j]
  const unsigned short* pi = prot73 + (size_t)(b * 384 + i) * 80;
  for (int idx = tid; idx < 64 * 73; idx += 256) {
    int row = idx / 73, f = idx - row * 73;
    Xs[row * 228 + f]      = pi[f];
    Xs[row * 228 + 73 + f] = prot73[(size_t)(b * 384 + j0 + row) * 80 + f];
  }
  for (int idx = tid; idx < 64 * 16; idx += 256) {  // cols 146..177: idx emb
    int row = idx >> 4, k = idx & 15;
    int sj = seq[b * 384 + j0 + row];
    float rel = (float)(si - sj);
    float denom = expf(((float)k * (1.f / 16.f)) * 7.6285176f);  // ln(2056)
    float ang = rel * 3.14159265358979f / denom;
    Xs[row * 228 + 146 + k] = f32_bf16(sinf(ang));
    Xs[row * 228 + 162 + k] = f32_bf16(cosf(ang));
  }
  for (int idx = tid; idx < 64 * 16; idx += 256) {  // cols 178..199: distogram
    int row = idx >> 4, sub = idx & 15;
    float dx = tix - trans[(b * 384 + j0 + row) * 3 + 0];
    float dy = tiy - trans[(b * 384 + j0 + row) * 3 + 1];
    float dz = tiz - trans[(b * 384 + j0 + row) * 3 + 2];
    float d = sqrtf(dx * dx + dy * dy + dz * dz);
    if (sub < 11) {
#pragma unroll
      for (int q = 0; q < 2; ++q) {
        int k = sub * 2 + q;
        float lo = 1e-5f + (float)k * ((20.f - 1e-5f) / 21.f);
        float hi = (k < 21) ? (1e-5f + (float)(k + 1) * ((20.f - 1e-5f) / 21.f)) : 1e8f;
        Xs[row * 228 + 178 + k] = f32_bf16((d > lo && d < hi) ? 1.f : 0.f);
      }
    }
  }
  for (int idx = tid; idx < 64 * 28; idx += 256) {  // zero pad 200..227
    int row = idx / 28, f = idx - row * 28;
    Xs[row * 228 + 200 + f] = 0;
  }
  __syncthreads();

  const int lane = tid & 31;
  const int wv   = tid >> 5;         // n-tile (0..7 -> 128 cols)
  const int ma   = lane & 15;
  const int kh   = lane >> 4;
  const int ncol = wv * 16 + ma;

  v8f acc[4];
#pragma unroll
  for (int mt = 0; mt < 4; ++mt) acc[mt] = (v8f){0.f,0.f,0.f,0.f,0.f,0.f,0.f,0.f};
  mm64<7>(Xw, SXW, w1f, wv, ma, kh, lane, acc);
  {
    float bb = b1[ncol];
#pragma unroll
    for (int mt = 0; mt < 4; ++mt) {
      v8f h;
#pragma unroll
      for (int v = 0; v < 8; ++v) h[v] = fmaxf(acc[mt][v] + bb, 0.f);
      store_h((unsigned short*)Hw0, 132, mt * 16, wv, kh, lane, h);
    }
  }
  __syncthreads();
#pragma unroll
  for (int mt = 0; mt < 4; ++mt) acc[mt] = (v8f){0.f,0.f,0.f,0.f,0.f,0.f,0.f,0.f};
  mm64<4>(Hw0, SHW, w2f, wv, ma, kh, lane, acc);
  {
    float bb = b2[ncol];
#pragma unroll
    for (int mt = 0; mt < 4; ++mt) {
      v8f h;
#pragma unroll
      for (int v = 0; v < 8; ++v) h[v] = fmaxf(acc[mt][v] + bb, 0.f);
      store_h((unsigned short*)Hw1, 132, mt * 16, wv, kh, lane, h);
    }
  }
  __syncthreads();
#pragma unroll
  for (int mt = 0; mt < 4; ++mt) acc[mt] = (v8f){0.f,0.f,0.f,0.f,0.f,0.f,0.f,0.f};
  mm64<4>(Hw1, SHW, w3f, wv, ma, kh, lane, acc);
  {
    float bb = b3[ncol];
#pragma unroll
    for (int mt = 0; mt < 4; ++mt)
#pragma unroll
      for (int v = 0; v < 8; ++v)
        H3[(mt * 16 + kh * 8 + v) * SH3 + ncol] = acc[mt][v] + bb;
  }
  __syncthreads();
  // LayerNorm over 128 per row; 16 lanes/row, 8 cols each; 4 row-groups.
#pragma unroll
  for (int it = 0; it < 4; ++it) {
    int row = it * 16 + (tid >> 4), c0 = (tid & 15) * 8;
    float vals[8]; float s = 0.f, ss = 0.f;
#pragma unroll
    for (int q = 0; q < 8; ++q) { float x = H3[row * SH3 + c0 + q]; vals[q] = x; s += x; ss += x * x; }
#pragma unroll
    for (int msk = 1; msk < 16; msk <<= 1) { s += __shfl_xor(s, msk); ss += __shfl_xor(ss, msk); }
    float mu = s * (1.f / 128.f);
    float var = ss * (1.f / 128.f) - mu * mu;
    float rs = rsqrtf(var + 1e-5f);
    float* po = out + (size_t)(r0 + row) * 128 + c0;
#pragma unroll
    for (int q = 0; q < 8; ++q) po[q] = (vals[q] - mu) * rs * g[c0 + q] + bt[c0 + q];
  }
}

// ---------------------------------------------------------------------------
// Node MLP: rows = B*N, NODE_IN=145 (pad 160) -> 256 -> 256 -> 256 + LN
// ---------------------------------------------------------------------------
__global__ __launch_bounds__(256) void node_mlp(
    const unsigned short* __restrict__ prot73, const int* __restrict__ seq,
    const float* __restrict__ trans, const float* __restrict__ rots,
    const float* __restrict__ at14,
    const unsigned* __restrict__ w1f, const unsigned* __restrict__ w2f,
    const unsigned* __restrict__ w3f,
    const float* __restrict__ b1, const float* __restrict__ b2,
    const float* __restrict__ b3,
    const float* __restrict__ g, const float* __restrict__ bt,
    float* __restrict__ out) {
  constexpr int SXW = 82;    // 164 bf16
  constexpr int SHW = 132;   // 264 bf16
  constexpr int SH3 = 260;
  __shared__ unsigned Xw[16 * SXW];
  __shared__ unsigned Hw[2][16 * SHW];
  __shared__ float H3[16 * SH3];
  unsigned short* Xs = (unsigned short*)Xw;
  const int tid = threadIdx.x;
  const int r0 = blockIdx.x * 16;

  for (int idx = tid; idx < 16 * 73; idx += 256) {
    int row = idx / 73, f = idx - row * 73;
    Xs[row * 164 + f] = prot73[(size_t)(r0 + row) * 80 + f];
  }
  {  // cols 73..104: index_embedding(seq_idx)
    int row = tid >> 4, k = tid & 15;
    float sv = (float)seq[r0 + row];
    float denom = expf(((float)k * (1.f / 16.f)) * 7.6285176f);
    float ang = sv * 3.14159265358979f / denom;
    Xs[row * 164 + 73 + k] = f32_bf16(sinf(ang));
    Xs[row * 164 + 89 + k] = f32_bf16(cosf(ang));
  }
  {  // cols 105..134 local, 135..144 1/(1+|local|)
    int row = tid >> 4, a = tid & 15;
    if (a < 10) {
      int gr = r0 + row;
      const float* R = rots + (size_t)gr * 9;
      const float* T = trans + (size_t)gr * 3;
      const float* A = at14 + (size_t)gr * 42 + (4 + a) * 3;
      float px = A[0] - T[0], py = A[1] - T[1], pz = A[2] - T[2];
      float l0 = R[0] * px + R[3] * py + R[6] * pz;   // einsum bnji,bnaj->bnai
      float l1 = R[1] * px + R[4] * py + R[7] * pz;
      float l2 = R[2] * px + R[5] * py + R[8] * pz;
      Xs[row * 164 + 105 + 3 * a + 0] = f32_bf16(l0);
      Xs[row * 164 + 105 + 3 * a + 1] = f32_bf16(l1);
      Xs[row * 164 + 105 + 3 * a + 2] = f32_bf16(l2);
      float d = sqrtf(l0 * l0 + l1 * l1 + l2 * l2);
      Xs[row * 164 + 135 + a] = f32_bf16(1.f / (1.f + d));
    }
  }
  for (int idx = tid; idx < 16 * 19; idx += 256) {  // zero pad 145..163
    int row = idx / 19, f = idx - row * 19;
    Xs[row * 164 + 145 + f] = 0;
  }
  __syncthreads();

  const int lane = tid & 31, wv = tid >> 5, ma = lane & 15, kh = lane >> 4;
  const int n0 = wv * 16 + ma, n1 = (wv + 8) * 16 + ma;

  v8f a0 = mm16<5>(Xw, SXW, w1f, wv,     ma, kh, lane);
  v8f a1 = mm16<5>(Xw, SXW, w1f, wv + 8, ma, kh, lane);
  {
    float bb0 = b1[n0], bb1 = b1[n1];
    v8f h0, h1;
#pragma unroll
    for (int v = 0; v < 8; ++v) { h0[v] = fmaxf(a0[v] + bb0, 0.f); h1[v] = fmaxf(a1[v] + bb1, 0.f); }
    store_h((unsigned short*)Hw[0], 264, 0, wv,     kh, lane, h0);
    store_h((unsigned short*)Hw[0], 264, 0, wv + 8, kh, lane, h1);
  }
  __syncthreads();
  a0 = mm16<8>(Hw[0], SHW, w2f, wv,     ma, kh, lane);
  a1 = mm16<8>(Hw[0], SHW, w2f, wv + 8, ma, kh, lane);
  {
    float bb0 = b2[n0], bb1 = b2[n1];
    v8f h0, h1;
#pragma unroll
    for (int v = 0; v < 8; ++v) { h0[v] = fmaxf(a0[v] + bb0, 0.f); h1[v] = fmaxf(a1[v] + bb1, 0.f); }
    store_h((unsigned short*)Hw[1], 264, 0, wv,     kh, lane, h0);
    store_h((unsigned short*)Hw[1], 264, 0, wv + 8, kh, lane, h1);
  }
  __syncthreads();
  a0 = mm16<8>(Hw[1], SHW, w3f, wv,     ma, kh, lane);
  a1 = mm16<8>(Hw[1], SHW, w3f, wv + 8, ma, kh, lane);
  {
    float bb0 = b3[n0], bb1 = b3[n1];
#pragma unroll
    for (int v = 0; v < 8; ++v) {
      H3[(kh * 8 + v) * SH3 + n0] = a0[v] + bb0;
      H3[(kh * 8 + v) * SH3 + n1] = a1[v] + bb1;
    }
  }
  __syncthreads();
  {  // LayerNorm over 256; 16 lanes/row, 16 cols each
    int row = tid >> 4, c0 = (tid & 15) * 16;
    float vals[16]; float s = 0.f, ss = 0.f;
#pragma unroll
    for (int q = 0; q < 16; ++q) { float x = H3[row * SH3 + c0 + q]; vals[q] = x; s += x; ss += x * x; }
#pragma unroll
    for (int msk = 1; msk < 16; msk <<= 1) { s += __shfl_xor(s, msk); ss += __shfl_xor(ss, msk); }
    float mu = s * (1.f / 256.f);
    float var = ss * (1.f / 256.f) - mu * mu;
    float rs = rsqrtf(var + 1e-5f);
    float* po = out + (size_t)(r0 + row) * 256 + c0;
#pragma unroll
    for (int q = 0; q < 16; ++q) po[q] = (vals[q] - mu) * rs * g[c0 + q] + bt[c0 + q];
  }
}

// ---------------------------------------------------------------------------
extern "C" void kernel_launch(void* const* d_in, const int* in_sizes, int n_in,
                              void* d_out, int out_size, void* d_ws, size_t ws_size,
                              hipStream_t stream) {
  const float* noised  = (const float*)d_in[0];
  const int*   seq     = (const int*)d_in[1];
  const float* tval    = (const float*)d_in[2];
  const float* fm      = (const float*)d_in[3];
  // d_in[4] node_mask unused by reference output
  const float* sc_trans = (const float*)d_in[5];
  const float* sc_rots  = (const float*)d_in[6];
  const float* sc_at14  = (const float*)d_in[7];
  const float* nw1 = (const float*)d_in[8];   const float* nb1 = (const float*)d_in[9];
  const float* nw2 = (const float*)d_in[10];  const float* nb2 = (const float*)d_in[11];
  const float* nw3 = (const float*)d_in[12];  const float* nb3 = (const float*)d_in[13];
  const float* ng  = (const float*)d_in[14];  const float* nbt = (const float*)d_in[15];
  const float* ew1 = (const float*)d_in[16];  const float* eb1 = (const float*)d_in[17];
  const float* ew2 = (const float*)d_in[18];  const float* eb2 = (const float*)d_in[19];
  const float* ew3 = (const float*)d_in[20];  const float* eb3 = (const float*)d_in[21];
  const float* eg  = (const float*)d_in[22];  const float* ebt = (const float*)d_in[23];

  // ws layout (~584 KB total):
  //   [0, 122880)      prot73 bf16 (768 rows x 80)
  //   [131072, ...)    weight fragments (u32)
  char* ws = (char*)d_ws;
  unsigned short* prot73 = (unsigned short*)ws;
  unsigned* nw1f = (unsigned*)(ws + 131072);       // 16t x 5c x 256 = 20480
  unsigned* nw2f = nw1f + 20480;                   // 16t x 8c x 256 = 32768
  unsigned* nw3f = nw2f + 32768;                   // 32768
  unsigned* ew1f = nw3f + 32768;                   // 8t x 7c x 256 = 14336
  unsigned* ew2f = ew1f + 14336;                   // 8t x 4c x 256 = 8192
  unsigned* ew3f = ew2f + 8192;                    // 8192

  prep_prot73_k<<<3, 256, 0, stream>>>(noised, tval, fm, prot73);
  prep_frags_k<<<80, 256, 0, stream>>>(nw1, 145, 256, 5, nw1f, 20480);
  prep_frags_k<<<128, 256, 0, stream>>>(nw2, 256, 256, 8, nw2f, 32768);
  prep_frags_k<<<128, 256, 0, stream>>>(nw3, 256, 256, 8, nw3f, 32768);
  prep_frags_k<<<56, 256, 0, stream>>>(ew1, 200, 128, 7, ew1f, 14336);
  prep_frags_k<<<32, 256, 0, stream>>>(ew2, 128, 128, 4, ew2f, 8192);
  prep_frags_k<<<32, 256, 0, stream>>>(ew3, 128, 128, 4, ew3f, 8192);

  float* out = (float*)d_out;
  node_mlp<<<48, 256, 0, stream>>>(prot73, seq, sc_trans, sc_rots, sc_at14,
                                   nw1f, nw2f, nw3f, nb1, nb2, nb3, ng, nbt, out);
  edge_mlp<<<2 * 384 * 384 / 64, 256, 0, stream>>>(prot73, seq, sc_trans,
                                                   ew1f, ew2f, ew3f, eb1, eb2, eb3,
                                                   eg, ebt, out + 2 * 384 * 256);
}
